// ActionExpertCrossBlock_78056735637554
// MI455X (gfx1250) — compile-verified, matrix operand order checked by reading, not run on hardware
//
#include <hip/hip_runtime.h>

// ---------- vector types matching CDNA5 WMMA operand shapes ----------
typedef _Float16 f16x16 __attribute__((ext_vector_type(16)));
typedef _Float16 f16x8  __attribute__((ext_vector_type(8)));
typedef float    f32x8  __attribute__((ext_vector_type(8)));
typedef int      vi4    __attribute__((vector_size(16)));   // async-copy payload type

static __device__ __forceinline__ f32x8 wmma32(f16x16 a, f16x16 b, f32x8 c) {
    return __builtin_amdgcn_wmma_f32_16x16x32_f16(false, a, false, b, (short)0, c, false, false);
}

// Load one 16-lane fragment slice of V_WMMA_F32_16X16X32_F16 (A or B).
// Caller passes p = base + row*ld + k0 + 8*hb (hb = lane>=16); the lane's 16
// halves are [p..p+8) and [p+16..p+24)  -> two b128 loads (global or ds).
static __device__ __forceinline__ f16x16 frag_ld(const _Float16* p) {
    f16x8 lo = *(const f16x8*)(p);
    f16x8 hi = *(const f16x8*)(p + 16);
    f16x16 f;
#pragma unroll
    for (int i = 0; i < 8; ++i) { f[i] = lo[i]; f[i + 8] = hi[i]; }
    return f;
}

// ---------- CDNA5 async global->LDS staging (ASYNCcnt-tracked DMA) ----------
static __device__ __forceinline__ void stage16(const _Float16* g, _Float16* l) {
#if __has_builtin(__builtin_amdgcn_global_load_async_to_lds_b128)
    __builtin_amdgcn_global_load_async_to_lds_b128(
        (__attribute__((address_space(1))) vi4*)(_Float16*)g,
        (__attribute__((address_space(3))) vi4*)l, 0, 0);
#else
    *(f16x8*)l = *(const f16x8*)g;          // synchronous fallback
#endif
}

static __device__ __forceinline__ void wait_async() {
#if __has_builtin(__builtin_amdgcn_s_wait_asynccnt)
    __builtin_amdgcn_s_wait_asynccnt(0);
#else
    asm volatile("s_wait_asynccnt 0x0" ::: "memory");
#endif
}

// LDS row padding: 40 halves (80B = 20 banks) -> conflict-free 16-lane frag reads
#define LP 40

// ---------------------------- cast f32 -> f16 ----------------------------
__global__ void cast_f16_kernel(const float* __restrict__ in, _Float16* __restrict__ out, int n) {
    int stride = gridDim.x * blockDim.x;
    for (int i = blockIdx.x * blockDim.x + threadIdx.x; i < n; i += stride)
        out[i] = (_Float16)in[i];
}

// ---------------------------- RMSNorm (D=1024) ----------------------------
__global__ __launch_bounds__(256) void rmsnorm_kernel(const float* __restrict__ x,
                                                      const float* __restrict__ w,
                                                      _Float16* __restrict__ out) {
    const int D = 1024;
    int row = blockIdx.x;
    int tid = threadIdx.x;
    const float* xr = x + (size_t)row * D;
    __shared__ float red[256];
    float s = 0.f;
#pragma unroll
    for (int i = tid; i < D; i += 256) { float v = xr[i]; s += v * v; }
    red[tid] = s;
    __syncthreads();
    for (int o = 128; o > 0; o >>= 1) {
        if (tid < o) red[tid] += red[tid + o];
        __syncthreads();
    }
    float rms = rsqrtf(red[0] * (1.f / D) + 1.1920928955078125e-7f);
#pragma unroll
    for (int i = tid; i < D; i += 256)
        out[(size_t)row * D + i] = (_Float16)(xr[i] * rms * w[i]);
}

// ------------------- WMMA GEMM: C[M,N] = A[M,K] @ W[N,K]^T -------------------
// 256 threads = 8 waves (2M x 4N), wave tile 32x64, block tile 64x256.
// Double-buffered async LDS staging of both tiles; fragments fed by ds_load_b128.
// MODE 0: f32 store; 1: f16 store; 2: f32 + residual.
template <int MODE>
__global__ __launch_bounds__(256) void gemm_wmma_kernel(
    const _Float16* __restrict__ A, const _Float16* __restrict__ W,
    void* __restrict__ Cout, const float* __restrict__ resid,
    int M, int N, int K) {
    __shared__ _Float16 As[2][64][LP];
    __shared__ _Float16 Bs[2][256][LP];

    int t = threadIdx.x;
    int lane = t & 31;
    int wave = t >> 5;
    int lr = lane & 15;
    int hb = (lane >> 4) & 1;
    int wm = wave >> 2, wn = wave & 3;
    int mblk = blockIdx.y * 64, nblk = blockIdx.x * 256;

    auto stage = [&](int buf, int k0) {
        {   // A tile 64x32: 256 x 16B segments, one per thread
            int row = t >> 2, seg = t & 3;
            stage16(A + (size_t)(mblk + row) * K + k0 + seg * 8, &As[buf][row][seg * 8]);
        }
#pragma unroll
        for (int i = 0; i < 4; ++i) {   // B tile 256x32: 1024 segments, 4 per thread
            int s2 = i * 256 + t;
            int row = s2 >> 2, seg = s2 & 3;
            stage16(W + (size_t)(nblk + row) * K + k0 + seg * 8, &Bs[buf][row][seg * 8]);
        }
    };

    f32x8 acc[2][4];
    f32x8 zero = {};
#pragma unroll
    for (int mt = 0; mt < 2; ++mt)
#pragma unroll
        for (int nt = 0; nt < 4; ++nt) acc[mt][nt] = zero;

    stage(0, 0);
    wait_async();
    __syncthreads();

    int nk = K >> 5;
    for (int kk = 0; kk < nk; ++kk) {
        int cur = kk & 1;
        if (kk + 1 < nk) stage(cur ^ 1, (kk + 1) * 32);   // overlap DMA with WMMA

        f16x16 af[2], bf[4];
#pragma unroll
        for (int mt = 0; mt < 2; ++mt)
            af[mt] = frag_ld(&As[cur][wm * 32 + 16 * mt + lr][8 * hb]);
#pragma unroll
        for (int nt = 0; nt < 4; ++nt)
            bf[nt] = frag_ld(&Bs[cur][wn * 64 + 16 * nt + lr][8 * hb]);
#pragma unroll
        for (int mt = 0; mt < 2; ++mt)
#pragma unroll
            for (int nt = 0; nt < 4; ++nt)
                acc[mt][nt] = wmma32(af[mt], bf[nt], acc[mt][nt]);

        wait_async();
        __syncthreads();
    }

#pragma unroll
    for (int mt = 0; mt < 2; ++mt)
#pragma unroll
        for (int nt = 0; nt < 4; ++nt)
#pragma unroll
            for (int j = 0; j < 8; ++j) {
                int r = mblk + wm * 32 + 16 * mt + j + 8 * hb;  // C layout row
                int c = nblk + wn * 64 + 16 * nt + lr;          // C layout col
                size_t idx = (size_t)r * N + c;
                float v = acc[mt][nt][j];
                if (MODE == 0) ((float*)Cout)[idx] = v;
                else if (MODE == 1) ((_Float16*)Cout)[idx] = (_Float16)v;
                else ((float*)Cout)[idx] = v + resid[idx];
            }
}

// ------------------- fused gate/up GEMM + SiLU epilogue -------------------
// block tile 64x128, wave tile 32x32, shared A fragments, async double buffer.
__global__ __launch_bounds__(256) void gateup_wmma_kernel(
    const _Float16* __restrict__ A, const _Float16* __restrict__ Wg,
    const _Float16* __restrict__ Wu, _Float16* __restrict__ out, int N, int K) {
    __shared__ _Float16 As[2][64][LP];
    __shared__ _Float16 Gs[2][128][LP];
    __shared__ _Float16 Us[2][128][LP];

    int t = threadIdx.x;
    int lane = t & 31;
    int wave = t >> 5;
    int lr = lane & 15;
    int hb = (lane >> 4) & 1;
    int wm = wave >> 2, wn = wave & 3;
    int mblk = blockIdx.y * 64, nblk = blockIdx.x * 128;

    auto stage = [&](int buf, int k0) {
        {   int row = t >> 2, seg = t & 3;
            stage16(A + (size_t)(mblk + row) * K + k0 + seg * 8, &As[buf][row][seg * 8]); }
#pragma unroll
        for (int i = 0; i < 2; ++i) {
            int s2 = i * 256 + t;
            int row = s2 >> 2, seg = s2 & 3;
            stage16(Wg + (size_t)(nblk + row) * K + k0 + seg * 8, &Gs[buf][row][seg * 8]);
            stage16(Wu + (size_t)(nblk + row) * K + k0 + seg * 8, &Us[buf][row][seg * 8]);
        }
    };

    f32x8 ag[2][2], au[2][2];
    f32x8 zero = {};
#pragma unroll
    for (int mt = 0; mt < 2; ++mt)
#pragma unroll
        for (int nt = 0; nt < 2; ++nt) { ag[mt][nt] = zero; au[mt][nt] = zero; }

    stage(0, 0);
    wait_async();
    __syncthreads();

    int nk = K >> 5;
    for (int kk = 0; kk < nk; ++kk) {
        int cur = kk & 1;
        if (kk + 1 < nk) stage(cur ^ 1, (kk + 1) * 32);

        f16x16 af[2], gf[2], uf[2];
#pragma unroll
        for (int mt = 0; mt < 2; ++mt)
            af[mt] = frag_ld(&As[cur][wm * 32 + 16 * mt + lr][8 * hb]);
#pragma unroll
        for (int nt = 0; nt < 2; ++nt) {
            gf[nt] = frag_ld(&Gs[cur][wn * 32 + 16 * nt + lr][8 * hb]);
            uf[nt] = frag_ld(&Us[cur][wn * 32 + 16 * nt + lr][8 * hb]);
        }
#pragma unroll
        for (int mt = 0; mt < 2; ++mt)
#pragma unroll
            for (int nt = 0; nt < 2; ++nt) {
                ag[mt][nt] = wmma32(af[mt], gf[nt], ag[mt][nt]);
                au[mt][nt] = wmma32(af[mt], uf[nt], au[mt][nt]);
            }

        wait_async();
        __syncthreads();
    }

#pragma unroll
    for (int mt = 0; mt < 2; ++mt)
#pragma unroll
        for (int nt = 0; nt < 2; ++nt)
#pragma unroll
            for (int j = 0; j < 8; ++j) {
                int r = mblk + wm * 32 + 16 * mt + j + 8 * hb;
                int c = nblk + wn * 32 + 16 * nt + lr;
                float g = ag[mt][nt][j];
                float u = au[mt][nt][j];
                float s = g * __builtin_amdgcn_rcpf(1.f + __expf(-g));  // fast SiLU
                out[(size_t)r * N + c] = (_Float16)(s * u);
            }
}

// ---------------------------- RoPE + scale + transpose ----------------------------
__global__ void rope_kernel(const _Float16* __restrict__ q, _Float16* __restrict__ qr) {
    int idx = blockIdx.x * 256 + threadIdx.x;   // B*L*8*128 = 4194304
    int d2 = idx & 127;
    int h = (idx >> 7) & 7;
    int l = (idx >> 10) & 511;
    int b = idx >> 19;
    if (b >= 8) return;
    size_t in_base = ((size_t)(b * 512 + l)) * 2048 + h * 256;
    float x1 = (float)q[in_base + d2];
    float x2 = (float)q[in_base + d2 + 128];
    float rad = (float)l * __powf(10000.f, -(float)d2 * (1.f / 128.f));
    float sn = __sinf(rad), cs = __cosf(rad);
    const float scale = 0.0625f;                // 1/sqrt(256)
    size_t ob = ((size_t)((b * 8 + h) * 512 + l)) * 256;
    qr[ob + d2]       = (_Float16)((x1 * cs - x2 * sn) * scale);
    qr[ob + d2 + 128] = (_Float16)((x2 * cs + x1 * sn) * scale);
}

// ---------------------------- flash attention ----------------------------
// grid (qtile=8, head=8, batch=8), block = 128 = 4 waves; wave owns 16 q rows.
// K chunk async-staged to LDS (shared by all 4 waves); V chunk manually
// transposed to LDS; P round-trips through LDS (C-layout -> A-layout).
__global__ __launch_bounds__(128) void attn_kernel(
    const _Float16* __restrict__ Q, const _Float16* __restrict__ Km,
    const _Float16* __restrict__ Vm, _Float16* __restrict__ ctx) {
    __shared__ _Float16 kS[32][264];            // [key][d], padded (stride 4 banks)
    __shared__ _Float16 vT[256][LP];            // [d][key] transposed V chunk
    __shared__ _Float16 Pl[4][16][LP];          // per-wave P tile

    int tid = threadIdx.x;
    int lane = tid & 31;
    int wave = tid >> 5;
    int lr = lane & 15;
    int hb = (lane >> 4) & 1;
    int qt = blockIdx.x, h = blockIdx.y, b = blockIdx.z;

    const _Float16* qbase = Q + ((size_t)(b * 8 + h) * 512) * 256;
    const _Float16* kbase = Km + (size_t)b * 2048 * 256;
    const _Float16* vbase = Vm + (size_t)b * 2048 * 256;
    int row0 = qt * 64 + wave * 16;

    f16x16 qf[8];
#pragma unroll
    for (int dc = 0; dc < 8; ++dc)
        qf[dc] = frag_ld(qbase + (size_t)(row0 + lr) * 256 + dc * 32 + 8 * hb);

    f32x8 acc[16];
    f32x8 zero = {};
#pragma unroll
    for (int tt = 0; tt < 16; ++tt) acc[tt] = zero;
    float m[8], lsum[8];
#pragma unroll
    for (int j = 0; j < 8; ++j) { m[j] = -3.0e38f; lsum[j] = 0.f; }

    for (int c0 = 0; c0 < 2048; c0 += 32) {
        __syncthreads();                        // LDS buffers free for reuse

        // async-stage K chunk 32x256: 1024 x 16B segments, 8 per thread
#pragma unroll
        for (int i = 0; i < 8; ++i) {
            int s2 = i * 128 + tid;
            int key = s2 >> 5, dseg = s2 & 31;
            stage16(kbase + (size_t)(c0 + key) * 256 + dseg * 8, &kS[key][dseg * 8]);
        }
        // stage V chunk transposed: vT[d][key]
#pragma unroll
        for (int p = 0; p < 8; ++p) {
            int idx = p * 128 + tid;            // (dblk 0..31, key 0..31)
            int key = idx & 31;
            int dblk = idx >> 5;
            f16x8 vv = *(const f16x8*)(vbase + (size_t)(c0 + key) * 256 + dblk * 8);
#pragma unroll
            for (int j = 0; j < 8; ++j) vT[dblk * 8 + j][key] = vv[j];
        }
        wait_async();
        __syncthreads();                        // kS + vT visible to all waves

        // S = Q . K^T (16 q rows x 32 keys), f32 accumulate
        f32x8 s0 = zero, s1 = zero;
#pragma unroll
        for (int dc = 0; dc < 8; ++dc) {
            f16x16 k0 = frag_ld(&kS[lr][dc * 32 + 8 * hb]);
            f16x16 k1 = frag_ld(&kS[16 + lr][dc * 32 + 8 * hb]);
            s0 = wmma32(qf[dc], k0, s0);
            s1 = wmma32(qf[dc], k1, s1);
        }

        // online softmax: row r = j + 8*hb spans 16 lanes of one acc vgpr
#pragma unroll
        for (int j = 0; j < 8; ++j) {
            float mx = fmaxf(s0[j], s1[j]);
#pragma unroll
            for (int o = 1; o < 16; o <<= 1) mx = fmaxf(mx, __shfl_xor(mx, o, 32));
            float mn = fmaxf(m[j], mx);
            float al = __expf(m[j] - mn);
            float p0 = __expf(s0[j] - mn);
            float p1 = __expf(s1[j] - mn);
            float rs = p0 + p1;
#pragma unroll
            for (int o = 1; o < 16; o <<= 1) rs += __shfl_xor(rs, o, 32);
            lsum[j] = lsum[j] * al + rs;
            m[j] = mn;
#pragma unroll
            for (int tt = 0; tt < 16; ++tt) acc[tt][j] *= al;
            Pl[wave][j + 8 * hb][lr] = (_Float16)p0;
            Pl[wave][j + 8 * hb][lr + 16] = (_Float16)p1;
        }
        __syncthreads();                        // Pl in A-layout readable

        // O += P(16x32) . V(32x256)
        f16x16 pf = frag_ld(&Pl[wave][lr][8 * hb]);
#pragma unroll
        for (int tt = 0; tt < 16; ++tt) {
            f16x16 bf = frag_ld(&vT[tt * 16 + lr][8 * hb]);
            acc[tt] = wmma32(pf, bf, acc[tt]);
        }
    }

    // epilogue: normalize with fast rcp, write ctx f16 [b][l][h*256+d]
#pragma unroll
    for (int j = 0; j < 8; ++j) {
        float inv = __builtin_amdgcn_rcpf(lsum[j]);
        int r = row0 + j + 8 * hb;
#pragma unroll
        for (int tt = 0; tt < 16; ++tt) {
            int d = tt * 16 + lr;
            ctx[((size_t)(b * 512 + r)) * 2048 + h * 256 + d] =
                (_Float16)(acc[tt][j] * inv);
        }
    }
}

// ---------------------------- host side ----------------------------
extern "C" void kernel_launch(void* const* d_in, const int* in_sizes, int n_in,
                              void* d_out, int out_size, void* d_ws, size_t ws_size,
                              hipStream_t stream) {
    (void)in_sizes; (void)n_in; (void)out_size; (void)ws_size;
    const float* x      = (const float*)d_in[0];   // [8,512,1024]
    const float* text_k = (const float*)d_in[1];   // [8,2048,256]
    const float* text_v = (const float*)d_in[2];
    const float* ln1    = (const float*)d_in[3];
    const float* ln2    = (const float*)d_in[4];
    const float* Wq     = (const float*)d_in[5];   // [2048,1024]
    const float* Wk     = (const float*)d_in[6];   // [256,256]
    const float* Wv     = (const float*)d_in[7];
    const float* Wo     = (const float*)d_in[8];   // [1024,2048]
    const float* Wg     = (const float*)d_in[9];   // [4096,1024]
    const float* Wu     = (const float*)d_in[10];
    const float* Wd     = (const float*)d_in[11];  // [1024,4096]

    char* w = (char*)d_ws;
    size_t off = 0;
    auto alloc = [&](size_t bytes) -> void* {
        void* p = w + off;
        off += (bytes + 255) & ~(size_t)255;
        return p;
    };
    const size_t M = 4096;                          // B*L
    _Float16* Wq16 = (_Float16*)alloc(2048 * 1024 * 2);
    _Float16* Wk16 = (_Float16*)alloc(256 * 256 * 2);
    _Float16* Wv16 = (_Float16*)alloc(256 * 256 * 2);
    _Float16* Wo16 = (_Float16*)alloc(1024 * 2048 * 2);
    _Float16* Wg16 = (_Float16*)alloc((size_t)4096 * 1024 * 2);
    _Float16* Wu16 = (_Float16*)alloc((size_t)4096 * 1024 * 2);
    _Float16* Wd16 = (_Float16*)alloc((size_t)1024 * 4096 * 2);
    _Float16* tk16 = (_Float16*)alloc((size_t)16384 * 256 * 2);
    _Float16* tv16 = (_Float16*)alloc((size_t)16384 * 256 * 2);
    _Float16* h1   = (_Float16*)alloc(M * 1024 * 2);
    _Float16* q16  = (_Float16*)alloc(M * 2048 * 2);
    _Float16* qr   = (_Float16*)alloc(M * 2048 * 2);
    _Float16* k16  = (_Float16*)alloc((size_t)16384 * 256 * 2);
    _Float16* v16  = (_Float16*)alloc((size_t)16384 * 256 * 2);
    _Float16* ctx  = (_Float16*)alloc(M * 2048 * 2);
    float*    x2   = (float*)alloc(M * 1024 * 4);
    _Float16* h2   = (_Float16*)alloc(M * 1024 * 2);
    _Float16* ffn  = (_Float16*)alloc(M * 4096 * 2);

    // 1) weight / input casts to f16
    cast_f16_kernel<<<1024, 256, 0, stream>>>(Wq, Wq16, 2048 * 1024);
    cast_f16_kernel<<<256, 256, 0, stream>>>(Wk, Wk16, 256 * 256);
    cast_f16_kernel<<<256, 256, 0, stream>>>(Wv, Wv16, 256 * 256);
    cast_f16_kernel<<<1024, 256, 0, stream>>>(Wo, Wo16, 1024 * 2048);
    cast_f16_kernel<<<2048, 256, 0, stream>>>(Wg, Wg16, 4096 * 1024);
    cast_f16_kernel<<<2048, 256, 0, stream>>>(Wu, Wu16, 4096 * 1024);
    cast_f16_kernel<<<2048, 256, 0, stream>>>(Wd, Wd16, 1024 * 4096);
    cast_f16_kernel<<<2048, 256, 0, stream>>>(text_k, tk16, 16384 * 256);
    cast_f16_kernel<<<2048, 256, 0, stream>>>(text_v, tv16, 16384 * 256);

    // 2) rmsnorm1
    rmsnorm_kernel<<<M, 256, 0, stream>>>(x, ln1, h1);

    // 3) Q projection: [4096,1024] @ Wq^T -> [4096,2048] f16
    gemm_wmma_kernel<1><<<dim3(2048 / 256, M / 64), 256, 0, stream>>>(
        h1, Wq16, q16, nullptr, M, 2048, 1024);

    // 4) RoPE + scale + [b][h][l][d] transpose
    rope_kernel<<<16384, 256, 0, stream>>>(q16, qr);

    // 5) K/V projections: [16384,256] @ W^T -> [16384,256] f16
    gemm_wmma_kernel<1><<<dim3(1, 16384 / 64), 256, 0, stream>>>(
        tk16, Wk16, k16, nullptr, 16384, 256, 256);
    gemm_wmma_kernel<1><<<dim3(1, 16384 / 64), 256, 0, stream>>>(
        tv16, Wv16, v16, nullptr, 16384, 256, 256);

    // 6) flash cross-attention -> ctx [b][l][2048] f16
    attn_kernel<<<dim3(8, 8, 8), 128, 0, stream>>>(qr, k16, v16, ctx);

    // 7) out projection + residual: ctx @ Wo^T + x -> x2 f32
    gemm_wmma_kernel<2><<<dim3(1024 / 256, M / 64), 256, 0, stream>>>(
        ctx, Wo16, x2, x, M, 1024, 2048);

    // 8) rmsnorm2
    rmsnorm_kernel<<<M, 256, 0, stream>>>(x2, ln2, h2);

    // 9) fused gate/up GEMM + SiLU -> ffn f16 [4096,4096]
    gateup_wmma_kernel<<<dim3(4096 / 128, M / 64), 256, 0, stream>>>(
        h2, Wg16, Wu16, ffn, 4096, 1024);

    // 10) down projection + residual -> d_out f32
    gemm_wmma_kernel<2><<<dim3(1024 / 256, M / 64), 256, 0, stream>>>(
        ffn, Wd16, (float*)d_out, x2, M, 1024, 4096);
}
